// DPSA3d_20916490731616
// MI455X (gfx1250) — matrix operand order repaired and checked by reading, not compiled, not run on hardware
//
#include <hip/hip_runtime.h>
#include <hip/hip_bf16.h>

#define NPOS 65536            // 16*64*64 voxels
#define CDIM 128
#define INNER 256             // HEADS*DIM_HEAD

typedef __attribute__((ext_vector_type(16))) _Float16 v16h;
typedef __attribute__((ext_vector_type(8)))  _Float16 v8h;
typedef __attribute__((ext_vector_type(4)))  _Float16 v4h;
typedef __attribute__((ext_vector_type(8)))  float    v8f;

// ---------------- WMMA fragment helpers (CDNA5 16x16x32 f16 layouts) ----------------

// A-matrix 16x32 from row-major tile (lda halves per row).
// lane group g=lane>>4: halves[0..7]=K(g*8..+7), halves[8..15]=K(16+g*8..+7), M=lane&15.
__device__ __forceinline__ v16h frag_a_row(const _Float16* t, int lda) {
  int lane = threadIdx.x & 31;
  int m = lane & 15, g = lane >> 4;
  v8h lo = *(const v8h*)(t + m * lda + g * 8);
  v8h hi = *(const v8h*)(t + m * lda + 16 + g * 8);
  v16h r;
#pragma unroll
  for (int i = 0; i < 8; ++i) { r[i] = lo[i]; r[i + 8] = hi[i]; }
  return r;
}

// B-matrix 32x16 from column-major tile ([n][k], k contiguous, ldb halves per column).
// lane group g: halves[0..7]=K(g*8..+7), halves[8..15]=K(16+g*8..+7), N = lane&15.
__device__ __forceinline__ v16h frag_b_col(const _Float16* t, int ldb) {
  int lane = threadIdx.x & 31;
  int n = lane & 15, g = lane >> 4;
  const _Float16* p = t + n * ldb + g * 8;
  v8h lo = *(const v8h*)(p);
  v8h hi = *(const v8h*)(p + 16);
  v16h r;
#pragma unroll
  for (int i = 0; i < 8; ++i) { r[i] = lo[i]; r[i + 8] = hi[i]; }
  return r;
}

#define WMMA_F16(A, B, C) \
  __builtin_amdgcn_wmma_f32_16x16x32_f16(false, (A), false, (B), (short)0, (C), false, false)

// async 16-byte global -> LDS copy (ASYNCcnt-tracked, CDNA5)
__device__ __forceinline__ void async_g2l_b128(unsigned lds_off, const void* gaddr) {
  asm volatile("global_load_async_to_lds_b128 %0, %1, off"
               :: "v"(lds_off), "v"((unsigned long long)(size_t)gaddr)
               : "memory");
}
__device__ __forceinline__ void wait_async0() {
  asm volatile("s_wait_asynccnt 0" ::: "memory");
}

// ---------------- 1) per-voxel LayerNorm stats over C=128 ----------------
__global__ __launch_bounds__(256) void k_ln_stats(const float* __restrict__ x,
                                                  float* __restrict__ mean,
                                                  float* __restrict__ rstd) {
  int p = blockIdx.x * 256 + threadIdx.x;
  float s = 0.f, s2 = 0.f;
#pragma unroll 4
  for (int c = 0; c < CDIM; ++c) {
    float v = x[(size_t)c * NPOS + p];
    s += v; s2 += v * v;
  }
  float m = s * (1.0f / CDIM);
  float var = s2 * (1.0f / CDIM) - m * m;
  mean[p] = m;
  rstd[p] = rsqrtf(var + 1e-5f);
}

// ---------------- 2) QKV GEMM: w_qkv(768x128) * xn ------------------------------
// q rows (o<256) -> qT[p][oc] (contiguous oc, b128 stores); k/v rows -> kv[c][p].
// grid = (NPOS/32, 768/64), block = 256 (8 waves: 4 M-subtiles x 2 N-subtiles)
__global__ __launch_bounds__(256) void k_qkv_gemm(const float* __restrict__ x,
                                                  const float* __restrict__ mean,
                                                  const float* __restrict__ rstd,
                                                  const float* __restrict__ g,
                                                  const float* __restrict__ b,
                                                  const float* __restrict__ wqkv,
                                                  _Float16* __restrict__ qT,
                                                  _Float16* __restrict__ kv) {
  __shared__ _Float16 As[64 * 32];   // [m][k] row-major
  __shared__ _Float16 Bt[32 * 32];   // [p][c] (B column-major)
  int tid = threadIdx.x;
  int p0 = blockIdx.x * 32;
  int m0 = blockIdx.y * 64;
  int wave = tid >> 5, lane = tid & 31;
  int wm = wave >> 1, wn = wave & 1;
  // staging index maps (vectorized: one v8h / v4h store per thread per chunk)
  int ami = tid >> 2, aki = (tid & 3) * 8;    // A: 64 rows x (4x8) halves
  int bpi = tid >> 3, bci = (tid & 7) * 4;    // B: 32 rows x (8x4) halves
  v8f acc = {};
  for (int kc = 0; kc < CDIM; kc += 32) {
    {   // A tile: 8 consecutive f32 -> one b128 LDS store
      const float* src = wqkv + (m0 + ami) * CDIM + kc + aki;
      v8h h8;
#pragma unroll
      for (int e = 0; e < 8; ++e) h8[e] = (_Float16)src[e];
      *(v8h*)&As[ami * 32 + aki] = h8;
    }
    {   // B tile: fused LayerNorm, 4 channels at one position -> b64 LDS store
      int p = p0 + bpi;
      float mp = mean[p], rp = rstd[p];
      v4h h4;
#pragma unroll
      for (int e = 0; e < 4; ++e) {
        int c = kc + bci + e;
        h4[e] = (_Float16)((x[(size_t)c * NPOS + p] - mp) * rp * g[c] + b[c]);
      }
      *(v4h*)&Bt[bpi * 32 + bci] = h4;
    }
    if (kc + 32 < CDIM) __builtin_prefetch(&x[(size_t)(kc + 32) * NPOS + p0 + tid], 0, 0);
    __syncthreads();
    v16h a  = frag_a_row(&As[(wm * 16) * 32], 32);
    v16h bb = frag_b_col(&Bt[(wn * 16) * 32], 32);
    acc = WMMA_F16(a, bb, acc);
    __syncthreads();
  }
  int n = lane & 15, mb = (lane >> 4) * 8;
  int ob = m0 + wm * 16 + mb;        // first of 8 consecutive output channels
  int p  = p0 + wn * 16 + n;
  if (m0 < INNER) {                  // q: contiguous channel octet at position p
    v8h h8;
#pragma unroll
    for (int r = 0; r < 8; ++r) h8[r] = (_Float16)acc[r];
    *(v8h*)(qT + (size_t)p * INNER + ob) = h8;
  } else {                           // k/v: keep [c][p] spatial-contiguous layout
#pragma unroll
    for (int r = 0; r < 8; ++r)
      kv[(size_t)(ob - INNER + r) * NPOS + p] = (_Float16)acc[r];
  }
}

// ---------------- 3) in-place channel L2 norm of q (qT) and k (kv) ----------------
__global__ __launch_bounds__(256) void k_l2norm(_Float16* __restrict__ qT,
                                                _Float16* __restrict__ kv) {
  int item = blockIdx.x * 256 + threadIdx.x;      // 8 heads * NPOS items
  int h = item >> 16, p = item & (NPOS - 1);
  // q: 32 contiguous halves
  {
    _Float16* base = qT + (size_t)p * INNER + h * 32;
    v8h v[4];
    float s = 0.f;
#pragma unroll
    for (int q4 = 0; q4 < 4; ++q4) {
      v[q4] = *(v8h*)(base + q4 * 8);
#pragma unroll
      for (int e = 0; e < 8; ++e) { float f = (float)v[q4][e]; s += f * f; }
    }
    float inv = 1.0f / fmaxf(sqrtf(s), 1e-12f);
#pragma unroll
    for (int q4 = 0; q4 < 4; ++q4) {
#pragma unroll
      for (int e = 0; e < 8; ++e) v[q4][e] = (_Float16)((float)v[q4][e] * inv);
      *(v8h*)(base + q4 * 8) = v[q4];
    }
  }
  // k: strided by NPOS
  {
    _Float16* base = kv + (size_t)(h * 32) * NPOS + p;
    float s = 0.f;
#pragma unroll
    for (int c = 0; c < 32; ++c) { float f = (float)base[(size_t)c * NPOS]; s += f * f; }
    float inv = 1.0f / fmaxf(sqrtf(s), 1e-12f);
#pragma unroll
    for (int c = 0; c < 32; ++c)
      base[(size_t)c * NPOS] = (_Float16)((float)base[(size_t)c * NPOS] * inv);
  }
}

// ---------------- 4) probes + top-k along D,H,W (one block per head) ---------------
__global__ __launch_bounds__(256) void k_topk(const _Float16* __restrict__ qT,
                                              const _Float16* __restrict__ kv,
                                              int* __restrict__ idxD,
                                              int* __restrict__ idxH,
                                              int* __restrict__ idxW) {
  __shared__ float qp[2048];
  __shared__ float kp[2048];
  __shared__ float score[64];
  __shared__ int selD[4];
  __shared__ int selH[8];
  int h = blockIdx.x, tid = threadIdx.x;
  const _Float16* qb = qT + h * 32;                       // [p][c] view
  const _Float16* kb = kv + (size_t)(h * 32) * NPOS;      // [c][p] view

  // ---- stage D ----
  for (int pair = tid; pair < 512; pair += 256) {
    int c = pair >> 4, d = pair & 15;
    const _Float16* kc = kb + (size_t)c * NPOS + d * 4096;
    float sq = 0.f, sk = 0.f;
    for (int i = 0; i < 4096; ++i) {
      sq += fabsf((float)qb[(size_t)(d * 4096 + i) * INNER + c]);
      sk += fabsf((float)kc[i]);
    }
    qp[pair] = sq; kp[pair] = sk;
  }
  __syncthreads();
  if (tid < 16) {
    float s = 0.f;
    for (int c = 0; c < 32; ++c) s += qp[c * 16 + tid] * kp[c * 16 + tid];
    score[tid] = s;
  }
  __syncthreads();
  if (tid == 0) {
    for (int r = 0; r < 4; ++r) {
      int best = 0; float bv = -1e30f;
      for (int d = 0; d < 16; ++d) if (score[d] > bv) { bv = score[d]; best = d; }
      score[best] = -1e30f; selD[r] = best; idxD[h * 4 + r] = best;
    }
  }
  __syncthreads();
  int d0 = selD[0], d1 = selD[1], d2 = selD[2], d3 = selD[3];
  __syncthreads();

  // ---- stage H ----
  for (int pair = tid; pair < 2048; pair += 256) {
    int c = pair >> 6, hh = pair & 63;
    const _Float16* kc = kb + (size_t)c * NPOS;
    float sq = 0.f, sk = 0.f;
    for (int d = 0; d < 16; ++d)
      for (int w = 0; w < 64; ++w)
        sq += fabsf((float)qb[(size_t)(d * 4096 + hh * 64 + w) * INNER + c]);
    int dd[4] = {d0, d1, d2, d3};
    for (int di = 0; di < 4; ++di)
      for (int w = 0; w < 64; ++w)
        sk += fabsf((float)kc[dd[di] * 4096 + hh * 64 + w]);
    qp[pair] = sq; kp[pair] = sk;
  }
  __syncthreads();
  if (tid < 64) {
    float s = 0.f;
    for (int c = 0; c < 32; ++c) s += qp[c * 64 + tid] * kp[c * 64 + tid];
    score[tid] = s;
  }
  __syncthreads();
  if (tid == 0) {
    for (int r = 0; r < 8; ++r) {
      int best = 0; float bv = -1e30f;
      for (int w = 0; w < 64; ++w) if (score[w] > bv) { bv = score[w]; best = w; }
      score[best] = -1e30f; selH[r] = best; idxH[h * 8 + r] = best;
    }
  }
  __syncthreads();

  // ---- stage W ----
  for (int pair = tid; pair < 2048; pair += 256) {
    int c = pair >> 6, w = pair & 63;
    const _Float16* kc = kb + (size_t)c * NPOS;
    float sq = 0.f, sk = 0.f;
    for (int d = 0; d < 16; ++d)
      for (int hh = 0; hh < 64; ++hh)
        sq += fabsf((float)qb[(size_t)(d * 4096 + hh * 64 + w) * INNER + c]);
    int dd[4] = {d0, d1, d2, d3};
    for (int di = 0; di < 4; ++di)
      for (int hi = 0; hi < 8; ++hi)
        sk += fabsf((float)kc[dd[di] * 4096 + selH[hi] * 64 + w]);
    qp[pair] = sq; kp[pair] = sk;
  }
  __syncthreads();
  if (tid < 64) {
    float s = 0.f;
    for (int c = 0; c < 32; ++c) s += qp[c * 64 + tid] * kp[c * 64 + tid];
    score[tid] = s;
  }
  __syncthreads();
  if (tid == 0) {
    for (int r = 0; r < 8; ++r) {
      int best = 0; float bv = -1e30f;
      for (int w = 0; w < 64; ++w) if (score[w] > bv) { bv = score[w]; best = w; }
      score[best] = -1e30f; idxW[h * 8 + r] = best;
    }
  }
}

// ---------------- 5) gather pruned K ([j][c], B-ready) and V^T ([c][j], B-ready) ----
__global__ __launch_bounds__(256) void k_gather(const _Float16* __restrict__ kv,
                                                const int* __restrict__ idxD,
                                                const int* __restrict__ idxH,
                                                const int* __restrict__ idxW,
                                                _Float16* __restrict__ kpr,
                                                _Float16* __restrict__ vprT) {
  int h = blockIdx.x, j = threadIdx.x;             // j = 0..255 pruned position
  int di = j >> 6, hi = (j >> 3) & 7, wi = j & 7;
  int p = idxD[h * 4 + di] * 4096 + idxH[h * 8 + hi] * 64 + idxW[h * 8 + wi];
  const _Float16* kb = kv + (size_t)(h * 32) * NPOS;
  const _Float16* vb = kv + (size_t)(INNER + h * 32) * NPOS;
#pragma unroll
  for (int c = 0; c < 32; ++c) {
    kpr [(size_t)(h * 256 + j) * 32 + c] = kb[(size_t)c * NPOS + p];
    vprT[(size_t)(h * 32 + c) * 256 + j] = vb[(size_t)c * NPOS + p];
  }
}

// ---------------- 6) attention: per head, per 16-query tile (1 wave / block) -------
__global__ __launch_bounds__(32) void k_attn(const _Float16* __restrict__ qT,
                                             const _Float16* __restrict__ kpr,
                                             const _Float16* __restrict__ vprT,
                                             _Float16* __restrict__ ofeatT) {
  __shared__ float    sim[16 * 256];     // scores
  __shared__ _Float16 at[16 * 256];      // attn weights f16 [i][j]
  int h = blockIdx.x >> 12;
  int p0 = (blockIdx.x & 4095) * 16;
  int lane = threadIdx.x;

  // q A-fragment straight from global: rows p0+i, 32 contiguous head-channels
  v16h aq = frag_a_row(qT + (size_t)p0 * INNER + h * 32, INNER);
  const _Float16* kb = kpr + (size_t)h * 256 * 32;
  int n = lane & 15, mb = (lane >> 4) * 8;
#pragma unroll
  for (int jt = 0; jt < 16; ++jt) {
    v16h bk = frag_b_col(kb + (size_t)jt * 16 * 32, 32);
    v8f c = {};
    c = WMMA_F16(aq, bk, c);
#pragma unroll
    for (int r = 0; r < 8; ++r) sim[(mb + r) * 256 + jt * 16 + n] = c[r];
  }
  __syncthreads();

  {   // rowwise softmax over 256 keys, 2 lanes per row, shfl_xor combine (wave32)
    int row = lane >> 1, half = lane & 1;
    float* srow = sim + row * 256;
    int j0 = half * 128;
    float mx = -1e30f;
    for (int j = j0; j < j0 + 128; ++j) mx = fmaxf(mx, srow[j]);
    mx = fmaxf(mx, __shfl_xor(mx, 1));
    float s = 0.f;
    for (int j = j0; j < j0 + 128; ++j) { float e = __expf(srow[j] - mx); srow[j] = e; s += e; }
    s += __shfl_xor(s, 1);
    float inv = 1.0f / s;
    for (int j = j0; j < j0 + 128; ++j) at[row * 256 + j] = (_Float16)(srow[j] * inv);
  }
  __syncthreads();

  const _Float16* vb = vprT + (size_t)h * 32 * 256;   // [c][j]
  v8f acc0 = {}, acc1 = {};
#pragma unroll
  for (int kc = 0; kc < 256; kc += 32) {
    v16h aa = frag_a_row(at + kc, 256);
    v16h b0 = frag_b_col(vb + kc, 256);                 // c = 0..15
    v16h b1 = frag_b_col(vb + 16 * 256 + kc, 256);      // c = 16..31
    acc0 = WMMA_F16(aa, b0, acc0);
    acc1 = WMMA_F16(aa, b1, acc1);
  }
  // write transposed output: ofeatT[p][oc] (B-ready for the output GEMM)
#pragma unroll
  for (int r = 0; r < 8; ++r) {
    int i = mb + r;
    ofeatT[(size_t)(p0 + i) * INNER + h * 32 + n]      = (_Float16)acc0[r];
    ofeatT[(size_t)(p0 + i) * INNER + h * 32 + 16 + n] = (_Float16)acc1[r];
  }
}

// ---------------- 7) output GEMM: y[128][NPOS] = w_out(128x256)*o + b_out ----------
// B tile staged with GLOBAL_LOAD_ASYNC_TO_LDS_B128 (ofeatT rows are already B-layout)
__global__ __launch_bounds__(256) void k_out_gemm(const float* __restrict__ wout,
                                                  const float* __restrict__ bout,
                                                  const _Float16* __restrict__ ofeatT,
                                                  float* __restrict__ y) {
  __shared__ _Float16 As[64 * 32];
  __shared__ _Float16 Bt[32 * 32];   // [p][oc], oc contiguous
  int tid = threadIdx.x;
  int p0 = blockIdx.x * 32;
  int m0 = blockIdx.y * 64;
  int wave = tid >> 5, lane = tid & 31;
  int wm = wave >> 1, wn = wave & 1;
  int ami = tid >> 2, aki = (tid & 3) * 8;    // A staging map
  v8f acc = {};
  for (int kc = 0; kc < INNER; kc += 32) {
    {   // A tile: 8 consecutive f32 -> one b128 LDS store
      const float* src = wout + (m0 + ami) * INNER + kc + aki;
      v8h h8;
#pragma unroll
      for (int e = 0; e < 8; ++e) h8[e] = (_Float16)src[e];
      *(v8h*)&As[ami * 32 + aki] = h8;
    }
    if (tid < 128) {                 // 128 lanes x 16B async copy: 32 rows x 64B
      int pi = tid >> 2, c8 = (tid & 3) * 8;
      async_g2l_b128((unsigned)(size_t)&Bt[pi * 32 + c8],
                     ofeatT + (size_t)(p0 + pi) * INNER + kc + c8);
    }
    wait_async0();
    __syncthreads();
    v16h a  = frag_a_row(&As[(wm * 16) * 32], 32);
    v16h bb = frag_b_col(&Bt[(wn * 16) * 32], 32);
    acc = WMMA_F16(a, bb, acc);
    __syncthreads();
  }
  int n = lane & 15, mb = (lane >> 4) * 8;
  int o = m0 + wm * 16 + mb;
  int p = p0 + wn * 16 + n;
#pragma unroll
  for (int r = 0; r < 8; ++r)
    y[(size_t)(o + r) * NPOS + p] = acc[r] + bout[o + r];
}

// ---------------- host-side launcher ----------------
extern "C" void kernel_launch(void* const* d_in, const int* in_sizes, int n_in,
                              void* d_out, int out_size, void* d_ws, size_t ws_size,
                              hipStream_t stream) {
  const float* x    = (const float*)d_in[0];
  const float* g    = (const float*)d_in[1];
  const float* b    = (const float*)d_in[2];
  const float* wqkv = (const float*)d_in[3];
  const float* wout = (const float*)d_in[4];
  const float* bout = (const float*)d_in[5];
  float* y = (float*)d_out;

  // workspace layout
  char* ws = (char*)d_ws;
  float* mean = (float*)ws;                               // NPOS f32
  float* rstd = mean + NPOS;                              // NPOS f32
  _Float16* qT   = (_Float16*)(rstd + NPOS);              // NPOS x 256 f16 (q, [p][oc])
  _Float16* kv   = qT + (size_t)NPOS * INNER;             // 512 x NPOS f16 (k,v [c][p])
  _Float16* kpr  = kv + (size_t)2 * INNER * NPOS;         // 8*256*32 f16
  _Float16* vprT = kpr + 8 * 256 * 32;                    // 8*32*256 f16
  _Float16* ofeatT = vprT + 8 * 32 * 256;                 // NPOS x 256 f16 ([p][oc])
  int* idxD = (int*)(ofeatT + (size_t)NPOS * INNER);      // 8*4
  int* idxH = idxD + 8 * 4;                               // 8*8
  int* idxW = idxH + 8 * 8;                               // 8*8

  k_ln_stats<<<NPOS / 256, 256, 0, stream>>>(x, mean, rstd);
  k_qkv_gemm<<<dim3(NPOS / 32, 768 / 64), 256, 0, stream>>>(x, mean, rstd, g, b, wqkv, qT, kv);
  k_l2norm<<<(8 * NPOS) / 256, 256, 0, stream>>>(qT, kv);
  k_topk<<<8, 256, 0, stream>>>(qT, kv, idxD, idxH, idxW);
  k_gather<<<8, 256, 0, stream>>>(kv, idxD, idxH, idxW, kpr, vprT);
  k_attn<<<8 * (NPOS / 16), 32, 0, stream>>>(qT, kpr, vprT, ofeatT);
  k_out_gemm<<<dim3(NPOS / 32, 128 / 64), 256, 0, stream>>>(wout, bout, ofeatT, y);
}
